// GraphFeaturesExtractor_81286551044800
// MI455X (gfx1250) — matrix operand order repaired and checked by reading, not compiled, not run on hardware
//
#include <hip/hip_runtime.h>
#include <math.h>

// ---------------------------------------------------------------------------
// 3-layer GAT for MI455X (gfx1250).
//   GEMMs via v_wmma_f32_16x16x32_f16 (f16 A/B, f32 accum) — dims all %16==0.
//   Double-buffered fragment pipeline so WMMAs overlap global loads
//   (partial s_wait_loadcnt instead of full drains).
//   Edge softmax/aggregation is L2-resident (feature matrix = 20MB << 192MB L2)
//   using fp32 global atomics.
// Workspace requirement: ~58 MB.
// ---------------------------------------------------------------------------

typedef __attribute__((ext_vector_type(16))) _Float16 v16h;
typedef __attribute__((ext_vector_type(8)))  float    v8f;
typedef __attribute__((ext_vector_type(4)))  unsigned int v4u;

#define F        512      // every layer's GEMM output width (HEADS*HID == OUT)
#define NTILES   32       // F / 16
#define SLOPE    0.2f

// ---------------- f32 -> f16 convert (node features for GEMM A) ------------
__global__ void k_f32_to_f16(const float* __restrict__ src,
                             _Float16* __restrict__ dst, int n)
{
    int i = blockIdx.x * blockDim.x + threadIdx.x;
    if (i < n) dst[i] = (_Float16)src[i];
}

// ---------------- pack W[K,F] (f32 row-major) into B-fragment layout -------
// Fragment for (kb, nt): lane L holds 16 contiguous halves =
//   W[kb*32 + (i>>3)*16 + (L>>4)*8 + (i&7)][nt*16 + (L&15)], i = 0..15
__global__ void k_pack_W(const float* __restrict__ W,
                         _Float16* __restrict__ Wp, int K)
{
    int i = blockIdx.x * blockDim.x + threadIdx.x;
    if (i >= K * F) return;
    int k = i / F, n = i % F;
    int kb = k >> 5, kr = k & 31;
    int lane = (((kr >> 3) & 1) << 4) | (n & 15);
    int idx  = (kr & 7) | ((kr >> 4) << 3);
    int nt   = n >> 4;
    size_t dst = ((((size_t)kb * NTILES + nt) * 32 + lane) << 4) + idx;
    Wp[dst] = (_Float16)W[(size_t)k * F + n];
}

// ---------------- WMMA GEMM: C[M,F] = A[M,K](f16) * W(packed f16) ----------
// One block (8 waves) per 16-row M tile; each wave produces 16x64 of C.
// Two-deep software pipeline on the K-block loop (KB is always even).
struct Frag { v4u u[2]; };

__device__ __forceinline__ v16h frag_h(const Frag& f) {
    union { Frag f; v16h h; } c;
    c.f = f;
    return c.h;
}

__device__ __forceinline__ void load_frags(const _Float16* __restrict__ arow,
                                           const _Float16* __restrict__ Wp,
                                           int wv, int lane, int kb,
                                           Frag& a, Frag b[4])
{
    a.u[0] = *(const v4u*)(arow + kb * 32);        // K = kb*32 + kA ..+7
    a.u[1] = *(const v4u*)(arow + kb * 32 + 16);   // K = kb*32 + kA+16..+23
    const _Float16* bbase =
        Wp + ((((size_t)kb * NTILES + wv * 4) * 32 + lane) << 4);
    #pragma unroll
    for (int t = 0; t < 4; ++t) {
        const v4u* bp = (const v4u*)(bbase + (size_t)t * 32 * 16);
        b[t].u[0] = bp[0];
        b[t].u[1] = bp[1];
    }
}

__device__ __forceinline__ void mma4(v8f acc[4], const Frag& a, const Frag b[4])
{
    #pragma unroll
    for (int t = 0; t < 4; ++t)
        acc[t] = __builtin_amdgcn_wmma_f32_16x16x32_f16(
                     false, frag_h(a), false, frag_h(b[t]),
                     (short)0, acc[t], false, false);
}

__global__ void __launch_bounds__(256)
k_gemm_wmma(const _Float16* __restrict__ A, const _Float16* __restrict__ Wp,
            float* __restrict__ C, int K)
{
    const int lane = threadIdx.x & 31;
    const int wv   = threadIdx.x >> 5;          // 0..7 -> n-tile group
    const int m0   = blockIdx.x * 16;
    const int mrow = lane & 15;
    const int kA   = (lane >> 4) * 8;           // A layout: lanes 16-31 hold K+8
    const _Float16* arow = A + (size_t)(m0 + mrow) * K + kA;

    v8f acc[4] = {};
    Frag a0, a1, b0[4], b1[4];
    const int KB = K >> 5;                      // 4 or 16 (even)

    load_frags(arow, Wp, wv, lane, 0, a0, b0);
    load_frags(arow, Wp, wv, lane, 1, a1, b1);

    int kb = 0;
    for (; kb < KB - 2; kb += 2) {
        __builtin_prefetch(arow + (kb + 4) * 32, 0, 3);   // A stream ahead
        mma4(acc, a0, b0);
        load_frags(arow, Wp, wv, lane, kb + 2, a0, b0);
        mma4(acc, a1, b1);
        load_frags(arow, Wp, wv, lane, kb + 3, a1, b1);
    }
    mma4(acc, a0, b0);
    mma4(acc, a1, b1);

    // C/D layout: VGPR r -> row m0 + (lane>=16 ? 8 : 0) + r, col = tile*16 + (lane&15)
    const int nb = wv * 64 + (lane & 15);
    const int mB = (lane >> 4) * 8;
    float* crow = C + (size_t)(m0 + mB) * F + nb;
    #pragma unroll
    for (int r = 0; r < 8; ++r) {
        crow[(size_t)r * F +  0] = acc[0][r];
        crow[(size_t)r * F + 16] = acc[1][r];
        crow[(size_t)r * F + 32] = acc[2][r];
        crow[(size_t)r * F + 48] = acc[3][r];
    }
}

// ---------------- attention logits: one wave per (node, head) --------------
__global__ void k_attn_logits(const float* __restrict__ hW,
                              const float* __restrict__ a_src,
                              const float* __restrict__ a_dst,
                              float* __restrict__ al_s, float* __restrict__ al_d,
                              int Nn, int H, int Cc)
{
    int wave = (int)((blockIdx.x * blockDim.x + threadIdx.x) >> 5);
    int lane = threadIdx.x & 31;
    if (wave >= Nn * H) return;                  // wave-uniform exit
    int n = wave / H, h = wave % H;
    const float* row = hW + (size_t)n * F + (size_t)h * Cc;
    const float* as  = a_src + (size_t)h * Cc;
    const float* ad  = a_dst + (size_t)h * Cc;
    float s = 0.f, d = 0.f;
    for (int c = lane; c < Cc; c += 32) {
        float v = row[c];
        s += v * as[c];
        d += v * ad[c];
    }
    #pragma unroll
    for (int off = 16; off > 0; off >>= 1) {
        s += __shfl_down(s, off, 32);
        d += __shfl_down(d, off, 32);
    }
    if (lane == 0) {
        al_s[(size_t)n * H + h] = s;
        al_d[(size_t)n * H + h] = d;
    }
}

// monotonic float <-> uint key for atomicMax over signed floats
__device__ __forceinline__ unsigned fkey(float x) {
    unsigned u = __float_as_uint(x);
    return (u & 0x80000000u) ? ~u : (u | 0x80000000u);
}
__device__ __forceinline__ float unfkey(unsigned k) {
    unsigned u = (k & 0x80000000u) ? (k & 0x7fffffffu) : ~k;
    return __uint_as_float(u);
}

__device__ __forceinline__ void edge_endpoints(const int* ei, int E, int e,
                                               int& src, int& dst)
{
    if (e < E) { src = ei[e]; dst = ei[E + e]; }     // edge_index rows 0/1
    else       { src = dst = e - E; }                 // appended self-loop
}

// ---------------- segment max (per dst,head) over leaky-relu logits --------
__global__ void k_edge_max(const int* __restrict__ ei, int E, int Nn, int H,
                           const float* __restrict__ al_s,
                           const float* __restrict__ al_d,
                           unsigned* __restrict__ smax)
{
    int i = blockIdx.x * blockDim.x + threadIdx.x;
    int Etot = E + Nn;
    if (i >= Etot * H) return;
    int e = i / H, h = i % H;
    int src, dst; edge_endpoints(ei, E, e, src, dst);
    float v = al_s[(size_t)src * H + h] + al_d[(size_t)dst * H + h];
    v = v > 0.f ? v : v * SLOPE;
    atomicMax(&smax[(size_t)dst * H + h], fkey(v));
}

// ---------------- exp(logit - max), segment sum ----------------------------
__global__ void k_edge_expsum(const int* __restrict__ ei, int E, int Nn, int H,
                              const float* __restrict__ al_s,
                              const float* __restrict__ al_d,
                              const unsigned* __restrict__ smax,
                              float* __restrict__ eexp, float* __restrict__ ssum)
{
    int i = blockIdx.x * blockDim.x + threadIdx.x;
    int Etot = E + Nn;
    if (i >= Etot * H) return;
    int e = i / H, h = i % H;
    int src, dst; edge_endpoints(ei, E, e, src, dst);
    float v = al_s[(size_t)src * H + h] + al_d[(size_t)dst * H + h];
    v = v > 0.f ? v : v * SLOPE;
    float m  = unfkey(smax[(size_t)dst * H + h]);
    float ex = __expf(v - m);
    eexp[(size_t)e * H + h] = ex;
    atomicAdd(&ssum[(size_t)dst * H + h], ex);
}

// ---------------- weighted aggregation: one wave per edge ------------------
__global__ void k_edge_aggregate(const int* __restrict__ ei, int E, int Nn,
                                 int H, int cshift,
                                 const float* __restrict__ hW,
                                 const float* __restrict__ eexp,
                                 const float* __restrict__ ssum,
                                 float* __restrict__ agg)
{
    int wave = (int)((blockIdx.x * blockDim.x + threadIdx.x) >> 5);
    int lane = threadIdx.x & 31;
    int Etot = E + Nn;
    if (wave >= Etot) return;                    // wave-uniform exit
    int src, dst; edge_endpoints(ei, E, wave, src, dst);
    float alph[4];
    for (int h = 0; h < H; ++h)
        alph[h] = eexp[(size_t)wave * H + h] / ssum[(size_t)dst * H + h];
    const float* hrow = hW  + (size_t)src * F;
    float*       orow = agg + (size_t)dst * F;
    for (int c = lane; c < F; c += 32)
        atomicAdd(&orow[c], hrow[c] * alph[c >> cshift]);
}

// ---------------- finalize: bias (+ relu) ----------------------------------
__global__ void k_bias_relu_f16(const float* __restrict__ agg,
                                const float* __restrict__ b,
                                _Float16* __restrict__ out, int n)
{
    int i = blockIdx.x * blockDim.x + threadIdx.x;
    if (i >= n) return;
    float v = agg[i] + b[i & (F - 1)];
    out[i] = (_Float16)(v > 0.f ? v : 0.f);
}

__global__ void k_bias_out(const float* __restrict__ agg,
                           const float* __restrict__ b,
                           float* __restrict__ out, int n)
{
    int i = blockIdx.x * blockDim.x + threadIdx.x;
    if (i >= n) return;
    out[i] = agg[i] + b[i & (F - 1)];
}

// ---------------------------------------------------------------------------
extern "C" void kernel_launch(void* const* d_in, const int* in_sizes, int n_in,
                              void* d_out, int out_size, void* d_ws, size_t ws_size,
                              hipStream_t stream)
{
    const float* x   = (const float*)d_in[0];
    const int*   ei  = (const int*)  d_in[1];
    const float* W[3]   = { (const float*)d_in[2], (const float*)d_in[6],  (const float*)d_in[10] };
    const float* asv[3] = { (const float*)d_in[3], (const float*)d_in[7],  (const float*)d_in[11] };
    const float* adv[3] = { (const float*)d_in[4], (const float*)d_in[8],  (const float*)d_in[12] };
    const float* bv[3]  = { (const float*)d_in[5], (const float*)d_in[9],  (const float*)d_in[13] };

    const int IN = 128;
    const int Nn = in_sizes[0] / IN;     // 10000
    const int E  = in_sizes[1] / 2;      // 320000
    const int Etot = E + Nn;

    // ---- workspace carve-out (256B aligned slices) ----
    char* ws = (char*)d_ws;
    size_t off = 0;
    auto take = [&](size_t bytes) -> void* {
        void* p = ws + off;
        off += (bytes + 255) & ~(size_t)255;
        return p;
    };
    _Float16* Af16 = (_Float16*)take((size_t)Nn * F * 2);   // GEMM A (f16)
    _Float16* Wp   = (_Float16*)take((size_t)F * F * 2);    // packed weights
    float*    hW   = (float*)   take((size_t)Nn * F * 4);   // x @ W
    float*    agg  = (float*)   take((size_t)Nn * F * 4);   // aggregated output
    float*    al_s = (float*)   take((size_t)Nn * 4 * 4);
    float*    al_d = (float*)   take((size_t)Nn * 4 * 4);
    unsigned* smax = (unsigned*)take((size_t)Nn * 4 * 4);
    float*    ssum = (float*)   take((size_t)Nn * 4 * 4);
    float*    eexp = (float*)   take((size_t)Etot * 4 * 4);

    // ---- one GAT layer ----
    auto layer = [&](int li, int K, int H, int Cc, int cshift, bool last) {
        // GEMM: hW = A(f16) @ W
        k_pack_W<<<(K * F + 255) / 256, 256, 0, stream>>>(W[li], Wp, K);
        k_gemm_wmma<<<Nn / 16, 256, 0, stream>>>(Af16, Wp, hW, K);
        // attention logits
        k_attn_logits<<<(Nn * H + 7) / 8, 256, 0, stream>>>(
            hW, asv[li], adv[li], al_s, al_d, Nn, H, Cc);
        // softmax state + output accumulator
        hipMemsetAsync(smax, 0, (size_t)Nn * H * 4, stream);
        hipMemsetAsync(ssum, 0, (size_t)Nn * H * 4, stream);
        hipMemsetAsync(agg,  0, (size_t)Nn * F * 4, stream);
        k_edge_max<<<(Etot * H + 255) / 256, 256, 0, stream>>>(
            ei, E, Nn, H, al_s, al_d, smax);
        k_edge_expsum<<<(Etot * H + 255) / 256, 256, 0, stream>>>(
            ei, E, Nn, H, al_s, al_d, smax, eexp, ssum);
        k_edge_aggregate<<<(Etot + 7) / 8, 256, 0, stream>>>(
            ei, E, Nn, H, cshift, hW, eexp, ssum, agg);
        // finalize
        if (!last)
            k_bias_relu_f16<<<(Nn * F + 255) / 256, 256, 0, stream>>>(
                agg, bv[li], Af16, Nn * F);
        else
            k_bias_out<<<(Nn * F + 255) / 256, 256, 0, stream>>>(
                agg, bv[li], (float*)d_out, Nn * F);
    };

    // input features -> f16
    k_f32_to_f16<<<(Nn * IN + 255) / 256, 256, 0, stream>>>(x, Af16, Nn * IN);

    layer(0, IN, 4, 128, 7, false);   // 128 -> 4x128, relu
    layer(1, F,  4, 128, 7, false);   // 512 -> 4x128, relu
    layer(2, F,  1, 512, 9, true);    // 512 -> 512, 1 head, no relu
}
